// MaskedMultiHeadAttention_15410342658603
// MI455X (gfx1250) — matrix operand order repaired and checked by reading, not compiled, not run on hardware
//
#include <hip/hip_runtime.h>
#include <stdint.h>

#define D_MODEL 1024
#define SEQ_LEN 2048
#define BATCH   2
#define NHEAD   16
#define HDIM    64
#define MTOT    (BATCH * SEQ_LEN)   // 4096

typedef unsigned short ushort_t;
typedef __bf16    bf16x16 __attribute__((ext_vector_type(16)));
typedef float     f32x8   __attribute__((ext_vector_type(8)));
typedef uint32_t  u32x4   __attribute__((ext_vector_type(4)));
typedef uint32_t  u32x8   __attribute__((ext_vector_type(8)));

union Frag { u32x4 q[2]; bf16x16 v; };

static __device__ __forceinline__ ushort_t f2bf(float f) {
  uint32_t u = __builtin_bit_cast(uint32_t, f);
  u += 0x7FFFu + ((u >> 16) & 1u);            // round-to-nearest-even
  return (ushort_t)(u >> 16);
}
static __device__ __forceinline__ float bf2f(ushort_t h) {
  uint32_t u = ((uint32_t)h) << 16;
  return __builtin_bit_cast(float, u);
}

// CDNA5 async copy: global -> LDS, tracked with ASYNCcnt (CDNA5 ISA)
static __device__ __forceinline__ void async_ld_b128(uint32_t lds_off, const void* g) {
  uint64_t ga = (uint64_t)g;
  asm volatile("global_load_async_to_lds_b128 %0, %1, off"
               :: "v"(lds_off), "v"(ga) : "memory");
}
static __device__ __forceinline__ void wait_async() {
  asm volatile("s_wait_asynccnt 0x0" ::: "memory");
}

static __device__ __forceinline__ f32x8 wmma_bf16(const Frag& a, const Frag& b, f32x8 c) {
  return __builtin_amdgcn_wmma_f32_16x16x32_bf16(false, a.v, false, b.v, (short)0, c, false, false);
}

// ---------------- fp32 -> bf16 conversion ----------------
__global__ void cvt_f32_bf16(const float* __restrict__ in, ushort_t* __restrict__ out, int n) {
  int i = blockIdx.x * blockDim.x + threadIdx.x;
  if (i < n) out[i] = f2bf(in[i]);
}

// ---------------- GEMM: C[M,N] = A[M,K] @ W[N,K]^T + bias ----------------
// 128 threads (4 waves). Block tile 128x64, K-step 32.
// Double-buffered LDS tiles filled with async-to-LDS copies (1 barrier / step).
// Each wave: 32 rows x 64 cols = 8 WMMAs per K-step (2 A-frags reuse 4 B-frags).
template <bool OUT_BF16>
__global__ void __launch_bounds__(128)
gemm_wmma(const ushort_t* __restrict__ A, const ushort_t* __restrict__ W,
          const float* __restrict__ bias, void* __restrict__ Cout) {
  const int K = D_MODEL, N = D_MODEL;
  __shared__ alignas(16) ushort_t As[2][128][32];   // [buf][row][k]
  __shared__ alignas(16) ushort_t Bs[2][64][32];    // [buf][n][k] (W rows == B^T tile)
  const int t = threadIdx.x;
  const int w = t >> 5, lane = t & 31;
  const int half = lane >> 4, n16 = lane & 15;
  const int nBase = blockIdx.x * 64, mBase = blockIdx.y * 128;

  f32x8 acc[2][4];
  for (int mi = 0; mi < 2; ++mi)
    for (int j = 0; j < 4; ++j)
      for (int r = 0; r < 8; ++r) acc[mi][j][r] = 0.0f;

  auto issue_tile = [&](int kk, int bufn) {
#pragma unroll
    for (int i = 0; i < 4; ++i) {                   // A: 128x32 bf16 = 512 x 16B
      int c = t + i * 128;
      int row = c >> 2, col = (c & 3) * 8;
      async_ld_b128((uint32_t)(uint64_t)(void*)&As[bufn][row][col],
                    &A[(size_t)(mBase + row) * K + kk + col]);
    }
#pragma unroll
    for (int i = 0; i < 2; ++i) {                   // B: 64x32 bf16 = 256 x 16B
      int c = t + i * 128;
      int row = c >> 2, col = (c & 3) * 8;
      async_ld_b128((uint32_t)(uint64_t)(void*)&Bs[bufn][row][col],
                    &W[(size_t)(nBase + row) * K + kk + col]);
    }
  };

  issue_tile(0, 0);
  int buf = 0;
  for (int ki = 0; ki < K / 32; ++ki) {
    wait_async();
    __syncthreads();
    if (ki + 1 < K / 32) issue_tile((ki + 1) * 32, buf ^ 1);

    const int kb = half * 8;
    Frag a0, a1;
    a0.q[0] = *(const u32x4*)&As[buf][w * 32 + n16][kb];
    a0.q[1] = *(const u32x4*)&As[buf][w * 32 + n16][kb + 16];
    a1.q[0] = *(const u32x4*)&As[buf][w * 32 + 16 + n16][kb];
    a1.q[1] = *(const u32x4*)&As[buf][w * 32 + 16 + n16][kb + 16];
    const int koff = half * 8;
#pragma unroll
    for (int j = 0; j < 4; ++j) {
      Frag b;
      b.q[0] = *(const u32x4*)&Bs[buf][j * 16 + n16][koff];
      b.q[1] = *(const u32x4*)&Bs[buf][j * 16 + n16][koff + 16];
      acc[0][j] = wmma_bf16(a0, b, acc[0][j]);
      acc[1][j] = wmma_bf16(a1, b, acc[1][j]);
    }
    buf ^= 1;
  }

#pragma unroll
  for (int mi = 0; mi < 2; ++mi)
#pragma unroll
    for (int j = 0; j < 4; ++j) {
      const int col = nBase + j * 16 + n16;
      const float bv = bias[col];
#pragma unroll
      for (int r = 0; r < 8; ++r) {
        const int row = mBase + w * 32 + mi * 16 + r + half * 8;
        const float v = acc[mi][j][r] + bv;
        if (OUT_BF16) ((ushort_t*)Cout)[(size_t)row * N + col] = f2bf(v);
        else          ((float*)Cout)[(size_t)row * N + col]    = v;
      }
    }
}

// ---------------- RoPE (in place on bf16 q,k) ----------------
__global__ void rope_kernel(ushort_t* __restrict__ q, ushort_t* __restrict__ k) {
  const int total = BATCH * SEQ_LEN * NHEAD * (HDIM / 2);
  int tid = blockIdx.x * blockDim.x + threadIdx.x;
  if (tid >= total) return;
  int pair = tid & 31;
  int tmp = tid >> 5;
  int h = tmp & 15;  tmp >>= 4;
  int s = tmp & 2047;
  int b = tmp >> 11;
  size_t off = ((size_t)(b * SEQ_LEN + s)) * D_MODEL + h * HDIM + pair * 2;
  float inv = __powf(10000.0f, -((float)(2 * pair)) / (float)HDIM);
  float ang = (float)s * inv;
  float c = __cosf(ang), sn = __sinf(ang);
  {
    float xe = bf2f(q[off]), xo = bf2f(q[off + 1]);
    q[off]     = f2bf(xe * c - xo * sn);
    q[off + 1] = f2bf(xe * sn + xo * c);
  }
  {
    float xe = bf2f(k[off]), xo = bf2f(k[off + 1]);
    k[off]     = f2bf(xe * c - xo * sn);
    k[off + 1] = f2bf(xe * sn + xo * c);
  }
}

// ---------------- causal flash attention (per b,h, 64 q-rows per block) ----------------
__global__ void __launch_bounds__(128)
attn_wmma(const ushort_t* __restrict__ Q, const ushort_t* __restrict__ Km,
          const ushort_t* __restrict__ V, ushort_t* __restrict__ O) {
  const int D = D_MODEL, S = SEQ_LEN;
  __shared__ alignas(16) ushort_t Ks[64][64];       // [key][d]   == B^T for QK^T
  __shared__ alignas(16) ushort_t Vt[64][64];       // [d][key]   == B^T for P@V
  __shared__ alignas(16) ushort_t Pl[4][16][64];    // per-wave P staging

  const int t = threadIdx.x, w = t >> 5, lane = t & 31;
  const int half = lane >> 4, n16 = lane & 15;
  const int qtile = blockIdx.x, h = blockIdx.y, b = blockIdx.z;
  const int qBase = qtile * 64;
  const size_t headOff = (size_t)h * HDIM;

  // Q fragments held in registers for the whole block (A layout, two K-steps)
  Frag q0, q1;
  {
    const int qrow = qBase + w * 16 + n16;
    const ushort_t* qp = Q + (size_t)(b * S + qrow) * D + headOff;
    const int kb = half * 8;
    q0.q[0] = *(const u32x4*)(qp + kb);
    q0.q[1] = *(const u32x4*)(qp + kb + 16);
    q1.q[0] = *(const u32x4*)(qp + 32 + kb);
    q1.q[1] = *(const u32x4*)(qp + 32 + kb + 16);
  }

  f32x8 Oa[4];
  float mrun[8], lrun[8];
  for (int j = 0; j < 4; ++j)
    for (int r = 0; r < 8; ++r) Oa[j][r] = 0.0f;
  for (int r = 0; r < 8; ++r) { mrun[r] = -__builtin_inff(); lrun[r] = 0.0f; }

  for (int kt = 0; kt <= qtile; ++kt) {
    const int kBase = kt * 64;

    // ---- K tile via Tensor Data Mover: 64x64 bf16 2D tile, row stride D ----
    // D# per CDNA5 ISA; wave 0 issues once for the WG
    // (TDM ignores EXEC; TENSORcnt is per-wave).
    if (w == 0) {
      uint64_t ga = (uint64_t)(Km + (size_t)(b * S + kBase) * D + headOff);
      u32x4 g0;
      g0[0] = 1u;                                   // count = 1 valid descriptor
      g0[1] = (uint32_t)(uint64_t)(void*)&Ks[0][0]; // lds_addr (bytes)
      g0[2] = (uint32_t)ga;                         // global_addr[31:0]
      g0[3] = ((uint32_t)(ga >> 32) & 0x01FFFFFFu)  // global_addr[56:32]
              | 0x80000000u;                        // type = 2 ("image")
      u32x8 g1;
      g1[0] = 1u << 16;                             // data_size = 1 -> 2 bytes
      g1[1] = 64u << 16;                            // tensor_dim0 = 64
      g1[2] = 64u << 16;                            // tensor_dim1 = 64 (dim0 hi=0)
      g1[3] = 64u << 16;                            // tile_dim0 = 64 (dim1 hi=0)
      g1[4] = 64u;                                  // tile_dim1 = 64, tile_dim2 = 0
      g1[5] = (uint32_t)D;                          // tensor_dim0_stride = 1024
      g1[6] = 0u;                                   // stride hi / dim1_stride lo
      g1[7] = 0u;
      asm volatile("tensor_load_to_lds %0, %1" :: "s"(g0), "s"(g1) : "memory");
      __builtin_amdgcn_s_wait_tensorcnt(0);
    }

    // ---- V tile: load + transpose into Vt[d][key] ----
#pragma unroll
    for (int i = 0; i < 4; ++i) {
      int c = t + i * 128;
      int key = c >> 3, p8 = (c & 7) * 8;
      const ushort_t* vp = V + (size_t)(b * S + kBase + key) * D + headOff + p8;
      u32x4 vv = *(const u32x4*)vp;
#pragma unroll
      for (int e = 0; e < 8; ++e) {
        uint32_t word = vv[e >> 1];
        Vt[p8 + e][key] = (ushort_t)((e & 1) ? (word >> 16) : (word & 0xffffu));
      }
    }
    __syncthreads();

    // S = (Q K^T) / sqrt(hd), causal mask on the diagonal tile
    f32x8 sc[4];
    const int koff = half * 8;
#pragma unroll
    for (int j = 0; j < 4; ++j) {
      Frag b0, b1;
      const int kr = j * 16 + n16;
      b0.q[0] = *(const u32x4*)&Ks[kr][koff];
      b0.q[1] = *(const u32x4*)&Ks[kr][koff + 16];
      b1.q[0] = *(const u32x4*)&Ks[kr][32 + koff];
      b1.q[1] = *(const u32x4*)&Ks[kr][32 + koff + 16];
      f32x8 z;
      for (int r = 0; r < 8; ++r) z[r] = 0.0f;
      z = wmma_bf16(q0, b0, z);
      z = wmma_bf16(q1, b1, z);
      sc[j] = z;
    }
#pragma unroll
    for (int j = 0; j < 4; ++j)
#pragma unroll
      for (int r = 0; r < 8; ++r) {
        float v = sc[j][r] * 0.125f;                 // 1/sqrt(64)
        if (kt == qtile) {
          int kidx = kBase + j * 16 + n16;
          int qidx = qBase + w * 16 + r + half * 8;
          if (kidx > qidx) v = -__builtin_inff();
        }
        sc[j][r] = v;
      }

    // online softmax (rows live across 16 lanes of each half-wave)
#pragma unroll
    for (int r = 0; r < 8; ++r) {
      float mt = fmaxf(fmaxf(sc[0][r], sc[1][r]), fmaxf(sc[2][r], sc[3][r]));
#pragma unroll
      for (int m = 1; m < 16; m <<= 1) mt = fmaxf(mt, __shfl_xor(mt, m, 32));
      float mnew = fmaxf(mrun[r], mt);
      float scale = __expf(mrun[r] - mnew);
      float psum = 0.0f;
#pragma unroll
      for (int j = 0; j < 4; ++j) {
        float p = __expf(sc[j][r] - mnew);
        sc[j][r] = p;
        psum += p;
      }
#pragma unroll
      for (int m = 1; m < 16; m <<= 1) psum += __shfl_xor(psum, m, 32);
      lrun[r] = lrun[r] * scale + psum;
      mrun[r] = mnew;
#pragma unroll
      for (int j = 0; j < 4; ++j) Oa[j][r] *= scale;
    }

    // stage P (C-layout -> LDS -> A-layout) for P@V
#pragma unroll
    for (int j = 0; j < 4; ++j)
#pragma unroll
      for (int r = 0; r < 8; ++r)
        Pl[w][r + half * 8][j * 16 + n16] = f2bf(sc[j][r]);
    __syncthreads();   // P visible; also: all waves done reading Ks

    Frag p0, p1;
    {
      const int kb = half * 8;
      p0.q[0] = *(const u32x4*)&Pl[w][n16][kb];
      p0.q[1] = *(const u32x4*)&Pl[w][n16][kb + 16];
      p1.q[0] = *(const u32x4*)&Pl[w][n16][32 + kb];
      p1.q[1] = *(const u32x4*)&Pl[w][n16][32 + kb + 16];
    }
#pragma unroll
    for (int j = 0; j < 4; ++j) {
      Frag b0, b1;
      const int dr = j * 16 + n16;
      b0.q[0] = *(const u32x4*)&Vt[dr][koff];
      b0.q[1] = *(const u32x4*)&Vt[dr][koff + 16];
      b1.q[0] = *(const u32x4*)&Vt[dr][32 + koff];
      b1.q[1] = *(const u32x4*)&Vt[dr][32 + koff + 16];
      Oa[j] = wmma_bf16(p0, b0, Oa[j]);
      Oa[j] = wmma_bf16(p1, b1, Oa[j]);
    }
    __syncthreads();   // all waves done with Vt/Pl/Ks before next tile load
  }

  // normalize and store bf16 head output
#pragma unroll
  for (int r = 0; r < 8; ++r) {
    const float inv = 1.0f / lrun[r];
    const int qrow = qBase + w * 16 + r + half * 8;
    ushort_t* op = O + (size_t)(b * S + qrow) * D + headOff;
#pragma unroll
    for (int j = 0; j < 4; ++j)
      op[j * 16 + n16] = f2bf(Oa[j][r] * inv);
  }
}

// ---------------- launch ----------------
extern "C" void kernel_launch(void* const* d_in, const int* in_sizes, int n_in,
                              void* d_out, int out_size, void* d_ws, size_t ws_size,
                              hipStream_t stream) {
  (void)in_sizes; (void)n_in; (void)out_size; (void)ws_size;
  const float* x  = (const float*)d_in[0];
  const float* Wq = (const float*)d_in[1];
  const float* bq = (const float*)d_in[2];
  const float* Wk = (const float*)d_in[3];
  const float* bk = (const float*)d_in[4];
  const float* Wv = (const float*)d_in[5];
  const float* bv = (const float*)d_in[6];
  const float* Wo = (const float*)d_in[7];
  const float* bo = (const float*)d_in[8];
  float* out = (float*)d_out;

  char* ws = (char*)d_ws;
  const size_t MB = 1024 * 1024;
  ushort_t* xb  = (ushort_t*)(ws);              // 8 MB  x  (bf16)
  ushort_t* Wqb = (ushort_t*)(ws + 8  * MB);    // 2 MB each weight
  ushort_t* Wkb = (ushort_t*)(ws + 10 * MB);
  ushort_t* Wvb = (ushort_t*)(ws + 12 * MB);
  ushort_t* Wob = (ushort_t*)(ws + 14 * MB);
  ushort_t* qb  = (ushort_t*)(ws + 16 * MB);    // 8 MB each
  ushort_t* kbf = (ushort_t*)(ws + 24 * MB);
  ushort_t* vb  = (ushort_t*)(ws + 32 * MB);
  ushort_t* ob  = (ushort_t*)(ws + 40 * MB);    // total 48 MB

  const int nx = MTOT * D_MODEL;        // 4M
  const int nw = D_MODEL * D_MODEL;     // 1M
  cvt_f32_bf16<<<(nx + 255) / 256, 256, 0, stream>>>(x,  xb,  nx);
  cvt_f32_bf16<<<(nw + 255) / 256, 256, 0, stream>>>(Wq, Wqb, nw);
  cvt_f32_bf16<<<(nw + 255) / 256, 256, 0, stream>>>(Wk, Wkb, nw);
  cvt_f32_bf16<<<(nw + 255) / 256, 256, 0, stream>>>(Wv, Wvb, nw);
  cvt_f32_bf16<<<(nw + 255) / 256, 256, 0, stream>>>(Wo, Wob, nw);

  dim3 gg(D_MODEL / 64, MTOT / 128);    // (16, 32)
  gemm_wmma<true><<<gg, 128, 0, stream>>>(xb, Wqb, bq, qb);
  gemm_wmma<true><<<gg, 128, 0, stream>>>(xb, Wkb, bk, kbf);
  gemm_wmma<true><<<gg, 128, 0, stream>>>(xb, Wvb, bv, vb);

  const int npair = BATCH * SEQ_LEN * NHEAD * (HDIM / 2);  // 2M
  rope_kernel<<<(npair + 255) / 256, 256, 0, stream>>>(qb, kbf);

  attn_wmma<<<dim3(SEQ_LEN / 64, NHEAD, BATCH), 128, 0, stream>>>(qb, kbf, vb, ob);

  gemm_wmma<false><<<gg, 128, 0, stream>>>(ob, Wob, bo, out);
}
